// WGNN_34282428957354
// MI455X (gfx1250) — matrix operand order repaired and checked by reading, not compiled, not run on hardware
//
#include <hip/hip_runtime.h>
#include <hip/hip_bf16.h>
#include <math.h>

typedef __attribute__((ext_vector_type(16))) _Float16 v16h;
typedef __attribute__((ext_vector_type(8)))  float    v8f;

#define NN_NODES 50000
#define NE_EDGES 1600000
#define IN_NODE  64
#define HID_NODE 32
#define IN_EDGE  8
#define EHID     80     // 10 * IN_EDGE
#define HEADS    4
#define CATW     128    // HEADS * HID_NODE

static __device__ __forceinline__ v8f wmma_f16(v16h a, v16h b, v8f c) {
  // D = A(16x32 f16) * B(32x16 f16) + C(16x16 f32)
  return __builtin_amdgcn_wmma_f32_16x16x32_f16(false, a, false, b, (short)0, c, false, false);
}

// ---------------------------------------------------------------------------
// Fused edge MLP: [E,8] -> ReLU(.@W1+b1)[80] -> ReLU(.@W2+b2)[80] -> .@W3+b3 [4]
// One wave processes 16 edges. Weight B-fragments prebuilt in LDS in WMMA
// per-lane layout (2x ds_load_b128 per fragment in the hot loop).
// ---------------------------------------------------------------------------
__global__ void __launch_bounds__(256)
edge_mlp_kernel(const float* __restrict__ EW,
                const float* __restrict__ W1, const float* __restrict__ B1,
                const float* __restrict__ W2, const float* __restrict__ B2,
                const float* __restrict__ W3, const float* __restrict__ B3,
                float* __restrict__ ewout, int E)
{
  // 23 B-fragments: [0..4]=L1 (K=8 pad 32, 5 N tiles), [5..19]=L2 (3 Kt x 5 Nt),
  // [20..22]=L3 (3 Kt, 1 Nt with 4 valid cols)
  __shared__ __attribute__((aligned(32))) _Float16 sB[23][32][16];     // 23 KB
  __shared__ __attribute__((aligned(16))) _Float16 sStage[8][16][96];  // 24 KB

  const int lane  = threadIdx.x & 31;
  const int wslot = threadIdx.x >> 5;

  // Prebuild B fragments (B layout: col = lane&15, K split {0..7,16..23}/{8..15,24..31} by lane group)
  for (int idx = threadIdx.x; idx < 23 * 32 * 16; idx += blockDim.x) {
    int i   = idx & 15;
    int ln  = (idx >> 4) & 31;
    int fid = idx >> 9;
    int krel = ((ln & 16) ? 8 : 0) + (i < 8 ? i : 8 + i);
    int ncol = ln & 15;
    float v = 0.f;
    if (fid < 5) {
      int n = fid * 16 + ncol;
      if (krel < IN_EDGE && n < EHID) v = W1[krel * EHID + n];
    } else if (fid < 20) {
      int f = fid - 5;
      int kt = f / 5;
      int n  = (f % 5) * 16 + ncol;
      int k  = kt * 32 + krel;
      if (k < EHID && n < EHID) v = W2[k * EHID + n];
    } else {
      int kt = fid - 20;
      int k  = kt * 32 + krel;
      if (k < EHID && ncol < HEADS) v = W3[k * HEADS + ncol];
    }
    sB[fid][ln][i] = (_Float16)v;
  }
  __syncthreads();

  _Float16 (*stage)[96] = sStage[wslot];
  // zero the K-pad columns (80..95) once; hot loop only writes cols 0..79
  if (lane < 16) {
    #pragma unroll
    for (int j = EHID; j < 96; ++j) stage[lane][j] = (_Float16)0.f;
  }

  const int m      = lane & 15;           // A row / output col index within tile
  const int hiHalf = (lane >> 4) & 1;
  const int koffA  = hiHalf ? 8 : 0;      // A-layout K split {0..7,16..23}/{8..15,24..31}

  // hoist loop-invariant biases (per-lane column is fixed)
  float b1v[5], b2v[5];
  #pragma unroll
  for (int nb = 0; nb < 5; ++nb) {
    b1v[nb] = B1[nb * 16 + m];
    b2v[nb] = B2[nb * 16 + m];
  }
  const float b3v = (m < HEADS) ? B3[m] : 0.f;

  const int nwaves = gridDim.x * (blockDim.x >> 5);
  const int wid    = blockIdx.x * (blockDim.x >> 5) + wslot;
  const int ntiles = (E + 15) >> 4;

  for (int t = wid; t < ntiles; t += nwaves) {
    const int e0 = t << 4;

    // ---- Layer 1: A = edge features 16x32 (only K<8 nonzero) ----
    v16h a1 = {};
    {
      int e = e0 + m;
      if (!hiHalf && e < E) {                 // lanes>=16 hold K>=8 -> all zero
        const float4* row = (const float4*)(EW + (size_t)e * IN_EDGE);
        float4 p0 = row[0];
        float4 p1 = row[1];
        a1[0] = (_Float16)p0.x; a1[1] = (_Float16)p0.y;
        a1[2] = (_Float16)p0.z; a1[3] = (_Float16)p0.w;
        a1[4] = (_Float16)p1.x; a1[5] = (_Float16)p1.y;
        a1[6] = (_Float16)p1.z; a1[7] = (_Float16)p1.w;
      }
    }
    v8f acc[5];
    #pragma unroll
    for (int n = 0; n < 5; ++n) {
      v8f z = {};
      acc[n] = wmma_f16(a1, *(const v16h*)sB[n][lane], z);
    }
    // relu(. + b1) -> staging LDS, f16, [16 rows][80 cols]
    #pragma unroll
    for (int n = 0; n < 5; ++n) {
      int col = n * 16 + m;
      #pragma unroll
      for (int r = 0; r < 8; ++r) {
        int rm = r + (hiHalf << 3);
        float v = acc[n][r] + b1v[n];
        stage[rm][col] = (_Float16)fmaxf(v, 0.f);
      }
    }

    // ---- Layer 2: A = 16x96 from staging (3 K tiles) ----
    v16h a2[3];
    #pragma unroll
    for (int kt = 0; kt < 3; ++kt) {
      #pragma unroll
      for (int i = 0; i < 16; ++i) {
        int k = kt * 32 + koffA + (i < 8 ? i : 8 + i);
        a2[kt][i] = stage[m][k];
      }
    }
    #pragma unroll
    for (int n = 0; n < 5; ++n) {
      v8f c = {};
      #pragma unroll
      for (int kt = 0; kt < 3; ++kt)
        c = wmma_f16(a2[kt], *(const v16h*)sB[5 + kt * 5 + n][lane], c);
      acc[n] = c;
    }
    #pragma unroll
    for (int n = 0; n < 5; ++n) {
      int col = n * 16 + m;
      #pragma unroll
      for (int r = 0; r < 8; ++r) {
        int rm = r + (hiHalf << 3);
        float v = acc[n][r] + b2v[n];
        stage[rm][col] = (_Float16)fmaxf(v, 0.f);
      }
    }

    // ---- Layer 3: 16x96 @ 96x16 (cols 0..3 valid), linear ----
    v16h a3[3];
    #pragma unroll
    for (int kt = 0; kt < 3; ++kt) {
      #pragma unroll
      for (int i = 0; i < 16; ++i) {
        int k = kt * 32 + koffA + (i < 8 ? i : 8 + i);
        a3[kt][i] = stage[m][k];
      }
    }
    v8f c3 = {};
    #pragma unroll
    for (int kt = 0; kt < 3; ++kt)
      c3 = wmma_f16(a3[kt], *(const v16h*)sB[20 + kt][lane], c3);

    if (m < HEADS) {
      #pragma unroll
      for (int r = 0; r < 8; ++r) {
        int e = e0 + r + (hiHalf << 3);
        if (e < E) ewout[(size_t)e * HEADS + m] = c3[r] + b3v;
      }
    }
  }
}

// ---------------------------------------------------------------------------
// Row-strip WMMA GEMM: one wave computes a 16-row strip across all N.
// B is prebuilt once per block in LDS in WMMA fragment layout
// (nkt * NTL fragments, <= 32). Per k-tile: 1 vectorized A load -> NTL WMMAs.
// act: 0=none, 1=relu, 2=elu(x)+1.001
// ---------------------------------------------------------------------------
template <int NTL>
__global__ void __launch_bounds__(256)
gemm_strip_kernel(const float* __restrict__ A, const float* __restrict__ B,
                  const float* __restrict__ bias, float* __restrict__ C,
                  int M, int N, int K, int act)
{
  const int nkt = (K + 31) >> 5;
  __shared__ __attribute__((aligned(32))) _Float16 sB[32][32][16];   // 32 KB max

  // Prebuild B fragments: fid = kt*NTL + nt
  for (int idx = threadIdx.x; idx < nkt * NTL * 512; idx += blockDim.x) {
    int i   = idx & 15;
    int ln  = (idx >> 4) & 31;
    int fid = idx >> 9;
    int kt  = fid / NTL, nt = fid % NTL;
    int k = kt * 32 + ((ln & 16) ? 8 : 0) + (i < 8 ? i : 8 + i);
    int n = nt * 16 + (ln & 15);
    float v = 0.f;
    if (k < K && n < N) v = B[(size_t)k * N + n];
    sB[fid][ln][i] = (_Float16)v;
  }
  __syncthreads();

  const int lane = threadIdx.x & 31;
  const int wid  = blockIdx.x * (blockDim.x >> 5) + (threadIdx.x >> 5);
  const int m0   = wid * 16;
  if (m0 >= M) return;                     // wave-uniform exit

  const int m      = m0 + (lane & 15);
  const int hiHalf = (lane >> 4) & 1;
  const int koffA  = hiHalf ? 8 : 0;
  const bool fastA = ((K & 31) == 0) && (m < M);

  v8f acc[NTL];
  #pragma unroll
  for (int t = 0; t < NTL; ++t) { v8f z = {}; acc[t] = z; }

  for (int kt = 0; kt < nkt; ++kt) {
    const int k0 = kt * 32;
    v16h a;
    if (fastA) {
      const float* ap = A + (size_t)m * K + k0 + koffA;   // 16B aligned
      float4 p0 = *(const float4*)(ap + 0);
      float4 p1 = *(const float4*)(ap + 4);
      float4 p2 = *(const float4*)(ap + 16);
      float4 p3 = *(const float4*)(ap + 20);
      a[0]  = (_Float16)p0.x; a[1]  = (_Float16)p0.y;
      a[2]  = (_Float16)p0.z; a[3]  = (_Float16)p0.w;
      a[4]  = (_Float16)p1.x; a[5]  = (_Float16)p1.y;
      a[6]  = (_Float16)p1.z; a[7]  = (_Float16)p1.w;
      a[8]  = (_Float16)p2.x; a[9]  = (_Float16)p2.y;
      a[10] = (_Float16)p2.z; a[11] = (_Float16)p2.w;
      a[12] = (_Float16)p3.x; a[13] = (_Float16)p3.y;
      a[14] = (_Float16)p3.z; a[15] = (_Float16)p3.w;
    } else {
      #pragma unroll
      for (int i = 0; i < 16; ++i) {
        int k = k0 + koffA + (i < 8 ? i : 8 + i);
        float v = 0.f;
        if (m < M && k < K) v = A[(size_t)m * K + k];
        a[i] = (_Float16)v;
      }
    }
    #pragma unroll
    for (int nt = 0; nt < NTL; ++nt)
      acc[nt] = wmma_f16(a, *(const v16h*)sB[kt * NTL + nt][lane], acc[nt]);
  }

  #pragma unroll
  for (int nt = 0; nt < NTL; ++nt) {
    int n = nt * 16 + (lane & 15);
    float bv = (bias != nullptr && n < N) ? bias[n] : 0.f;
    #pragma unroll
    for (int r = 0; r < 8; ++r) {
      int mr = m0 + r + (hiHalf << 3);
      if (mr < M && n < N) {
        float v = acc[nt][r] + bv;
        if (act == 1) v = fmaxf(v, 0.f);
        else if (act == 2) v = (v > 0.f ? v : (expf(v) - 1.f)) + 1.001f;
        C[(size_t)mr * N + n] = v;
      }
    }
  }
}

// ---------------------------------------------------------------------------
// Small helper kernels
// ---------------------------------------------------------------------------
__global__ void zero_kernel(float* __restrict__ p, int n) {
  int i = blockIdx.x * blockDim.x + threadIdx.x;
  if (i < n) p[i] = 0.f;
}

// conv_W [C, in, 32] -> Wcat [in, C*32]
__global__ void repack_kernel(const float* __restrict__ W, float* __restrict__ Wcat, int in_dim) {
  int i = blockIdx.x * blockDim.x + threadIdx.x;
  int total = in_dim * CATW;
  if (i >= total) return;
  int row = i >> 7;
  int f = i & 127;
  int c = f >> 5, o = f & 31;
  Wcat[i] = W[((size_t)c * in_dim + row) * HID_NODE + o];
}

// agg[dst] += h[src] * ew[e, head] ; 32 lanes per edge, 4 floats/lane
__global__ void scatter_kernel(const float* __restrict__ h, const float* __restrict__ ew,
                               const int* __restrict__ src, const int* __restrict__ dst,
                               float* __restrict__ agg, int E)
{
  long long t = (long long)blockIdx.x * blockDim.x + threadIdx.x;
  int e = (int)(t >> 5);
  if (e >= E) return;
  int l = (int)(t & 31);
  int f = l << 2;
  float w = ew[(size_t)e * HEADS + (f >> 5)];
  int s = src[e], d = dst[e];
  const float4 v = *(const float4*)(h + (size_t)s * CATW + f);
  float* o = agg + (size_t)d * CATW + f;
  atomicAdd(o + 0, v.x * w);
  atomicAdd(o + 1, v.y * w);
  atomicAdd(o + 2, v.z * w);
  atomicAdd(o + 3, v.w * w);
}

__global__ void bias_relu_kernel(const float* __restrict__ agg, const float* __restrict__ b,
                                 float* __restrict__ out, int total)
{
  int i = blockIdx.x * blockDim.x + threadIdx.x;
  if (i < total) {
    float v = agg[i] + b[i & 127];   // conv_b [4,32] flat == feature index c*32+o
    out[i] = fmaxf(v, 0.f);
  }
}

// ---------------------------------------------------------------------------
extern "C" void kernel_launch(void* const* d_in, const int* in_sizes, int n_in,
                              void* d_out, int out_size, void* d_ws, size_t ws_size,
                              hipStream_t stream)
{
  const float* x0   = (const float*)d_in[0];
  const int*   ei   = (const int*)d_in[1];      // edge_index [2,E]
  const float* ewin = (const float*)d_in[2];    // edge_weight [E,8]
  const float* eW1  = (const float*)d_in[3];
  const float* eB1  = (const float*)d_in[4];
  const float* eW2  = (const float*)d_in[5];
  const float* eB2  = (const float*)d_in[6];
  const float* eW3  = (const float*)d_in[7];
  const float* eB3  = (const float*)d_in[8];
  const float* cW0  = (const float*)d_in[9];
  const float* cW1  = (const float*)d_in[10];
  const float* cW2  = (const float*)d_in[11];
  const float* cB0  = (const float*)d_in[12];
  const float* cB1  = (const float*)d_in[13];
  const float* cB2  = (const float*)d_in[14];
  const float* rW1  = (const float*)d_in[15];
  const float* rB1  = (const float*)d_in[16];
  const float* rW2  = (const float*)d_in[17];
  const float* rB2  = (const float*)d_in[18];
  const float* rW3  = (const float*)d_in[19];
  const float* rB3  = (const float*)d_in[20];

  const int E = NE_EDGES, NN = NN_NODES;
  const int* srcp = ei;
  const int* dstp = ei + E;

  float* ws   = (float*)d_ws;
  float* ew   = ws;                                // E*4
  float* h    = ew   + (size_t)E * HEADS;          // NN*128
  float* agg  = h    + (size_t)NN * CATW;          // NN*128
  float* xbuf = agg  + (size_t)NN * CATW;          // NN*128
  float* wc0  = xbuf + (size_t)NN * CATW;          // 64*128
  float* wc1  = wc0  + (size_t)IN_NODE * CATW;     // 128*128
  float* wc2  = wc1  + (size_t)CATW * CATW;        // 128*128

  // repack conv weights to [in, 128]
  {
    int t0 = IN_NODE * CATW, t1 = CATW * CATW;
    repack_kernel<<<(t0 + 255) / 256, 256, 0, stream>>>(cW0, wc0, IN_NODE);
    repack_kernel<<<(t1 + 255) / 256, 256, 0, stream>>>(cW1, wc1, CATW);
    repack_kernel<<<(t1 + 255) / 256, 256, 0, stream>>>(cW2, wc2, CATW);
  }

  // edge MLP (fused WMMA) -> ew [E,4]
  edge_mlp_kernel<<<1024, 256, 0, stream>>>(ewin, eW1, eB1, eW2, eB2, eW3, eB3, ew, E);

  // 3 GCN layers: h = x @ Wcat ; agg = scatter(h*ew) ; x = relu(agg + b)
  const float* xin = x0;
  const float* wcs[3] = {wc0, wc1, wc2};
  const float* cbs[3] = {cB0, cB1, cB2};
  int kdims[3] = {IN_NODE, CATW, CATW};
  const int totalNC = NN * CATW;
  const int strips  = (NN + 15) / 16;              // 3125 waves
  const int sblocks = (strips + 7) / 8;
  for (int l = 0; l < 3; ++l) {
    gemm_strip_kernel<8><<<sblocks, 256, 0, stream>>>(xin, wcs[l], nullptr, h, NN, CATW, kdims[l], 0);
    zero_kernel<<<(totalNC + 255) / 256, 256, 0, stream>>>(agg, totalNC);
    long long thr = (long long)E * 32;
    scatter_kernel<<<(int)((thr + 255) / 256), 256, 0, stream>>>(h, ew, srcp, dstp, agg, E);
    bias_relu_kernel<<<(totalNC + 255) / 256, 256, 0, stream>>>(agg, cbs[l], xbuf, totalNC);
    xin = xbuf;
  }

  // readout MLP: 128 -> 128 (relu) -> 128 (relu) -> 1, then elu(x)+1.001
  gemm_strip_kernel<8><<<sblocks, 256, 0, stream>>>(xin, rW1, rB1, h,   NN, CATW, CATW, 1);
  gemm_strip_kernel<8><<<sblocks, 256, 0, stream>>>(h,   rW2, rB2, agg, NN, CATW, CATW, 1);
  gemm_strip_kernel<1><<<sblocks, 256, 0, stream>>>(agg, rW3, rB3, (float*)d_out, NN, 1, CATW, 2);

  (void)in_sizes; (void)n_in; (void)out_size; (void)ws_size;
}
